// ViT_13735305413385
// MI455X (gfx1250) — matrix operand order repaired and checked by reading, not compiled, not run on hardware
//
#include <hip/hip_runtime.h>

typedef __attribute__((ext_vector_type(16))) _Float16 v16h;
typedef __attribute__((ext_vector_type(8)))  _Float16 v8h;
typedef __attribute__((ext_vector_type(8)))  float    v8f;
typedef int v4i_vs __attribute__((vector_size(16)));  // pointee type of async-LDS builtins

// ---------------------------------------------------------------------------
// CDNA5 async global->LDS copies (ASYNCcnt path), with synchronous fallback.
// ---------------------------------------------------------------------------
#if defined(__gfx1250__) && __has_builtin(__builtin_amdgcn_global_load_async_to_lds_b128)
#define HAVE_ASYNC_LDS 1
#else
#define HAVE_ASYNC_LDS 0
#endif

__device__ __forceinline__ void async_copy16(const void* g, void* l) {
#if HAVE_ASYNC_LDS
  __builtin_amdgcn_global_load_async_to_lds_b128(
      (__attribute__((address_space(1))) v4i_vs*)(g),
      (__attribute__((address_space(3))) v4i_vs*)(l), 0, 0);
#else
  (void)g; (void)l;
#endif
}

__device__ __forceinline__ void async_wait0() {
#if HAVE_ASYNC_LDS
#if __has_builtin(__builtin_amdgcn_s_wait_asynccnt)
  __builtin_amdgcn_s_wait_asynccnt(0);
#else
  asm volatile("s_wait_asynccnt 0" ::: "memory");
#endif
#endif
}

// ---------------------------------------------------------------------------
// WMMA fragment loaders for v_wmma_f32_16x16x32_f16 (wave32), ISA 7.12.2.
// A tile: 16x32 row-major; lane m=lane&15, hl=lane>>4:
//   elems 0..15 <- A[m][hl*16 + 0..15]
// B tile stored N-major (transposed): Bt[n][k]; elems 0..15 <- Bt[n][hl*16+0..15]
// C/D: elem r -> row r + 8*hl, col lane&15.
// ---------------------------------------------------------------------------
__device__ __forceinline__ v16h load_a_frag(const _Float16* A, int lda) {
  int lane = threadIdx.x & 31;
  int m = lane & 15, hl = lane >> 4;
  const _Float16* r0 = A + m * lda + hl * 16;
  v8h lo = *(const v8h*)(r0);
  v8h hi = *(const v8h*)(r0 + 8);
  v16h a;
#pragma unroll
  for (int j = 0; j < 8; ++j) { a[j] = lo[j]; a[8 + j] = hi[j]; }
  return a;
}

__device__ __forceinline__ v16h load_b_frag_t(const _Float16* Bt, int ldb) {
  int lane = threadIdx.x & 31;
  int n = lane & 15, hl = lane >> 4;
  return *(const v16h*)(Bt + n * ldb + hl * 16);
}

#define WMMA_F16(a, b, c) \
  __builtin_amdgcn_wmma_f32_16x16x32_f16(false, (a), false, (b), (short)0, (c), false, false)

// ---------------------------------------------------------------------------
// Kernel 0: one-time f32 -> f16 weight conversion (enables async staging)
// ---------------------------------------------------------------------------
__global__ __launch_bounds__(256) void cvt_f16_kernel(const float* __restrict__ s,
                                                      _Float16* __restrict__ d, int n) {
  int i = blockIdx.x * 256 + threadIdx.x;
  if (i < n) d[i] = (_Float16)s[i];
}

__global__ __launch_bounds__(256) void cvt_pad_wh_kernel(const float* __restrict__ s,
                                                         _Float16* __restrict__ d) {
  int i = blockIdx.x * 256 + threadIdx.x;  // grid covers 1024*1024
  d[i] = (i < 1024000) ? (_Float16)s[i] : (_Float16)0.f;  // pad rows 1000..1023
}

// ---------------------------------------------------------------------------
// Kernel 1: patchify + patch embedding + cls token + sinusoidal pos emb
// ---------------------------------------------------------------------------
__device__ __forceinline__ float pos_emb(int p, int i) {
  float fp = (float)p;
  if ((i & 1) == 0) return __sinf(fp * __powf(10000.f, -(float)i * (1.f / 1024.f)));
  return __cosf(fp * __powf(10000.f, -(float)(i - 1) * (1.f / 1024.f)));
}

__global__ __launch_bounds__(256) void embed_kernel(const float* __restrict__ x,
                                                    const float* __restrict__ cls,
                                                    const float* __restrict__ Wp,
                                                    float* __restrict__ tok) {
  int t = blockIdx.x, bb = blockIdx.y;
  __shared__ float patch[16];
  if (t > 0 && threadIdx.x < 16) {
    int pp = t - 1, py = pp / 7, px = pp % 7;
    int r = threadIdx.x >> 2, c = threadIdx.x & 3;
    patch[threadIdx.x] = x[bb * 784 + (py * 4 + r) * 28 + (px * 4 + c)];
  }
  __syncthreads();
  float* out = tok + (long)(bb * 50 + t) * 1024;
  for (int d = threadIdx.x; d < 1024; d += 256) {
    float v;
    if (t == 0) {
      v = cls[d];
    } else {
      v = 0.f;
#pragma unroll
      for (int s = 0; s < 16; ++s) v += patch[s] * Wp[d * 16 + s];
    }
    out[d] = v + pos_emb(t, d);
  }
}

// ---------------------------------------------------------------------------
// Kernel 2: LayerNorm over D=1024, f16 output (feeds WMMA A operands)
// ---------------------------------------------------------------------------
__global__ __launch_bounds__(256) void ln_kernel(const float* __restrict__ src, int srow,
                                                 const float* __restrict__ w,
                                                 const float* __restrict__ b,
                                                 _Float16* __restrict__ dst, int drow) {
  long row = blockIdx.x;
  const float* xp = src + row * (long)srow;
  _Float16* op = dst + row * (long)drow;
  float vals[4];
  float lsum = 0.f, lsq = 0.f;
#pragma unroll
  for (int i = 0; i < 4; ++i) {
    float v = xp[threadIdx.x + 256 * i];
    vals[i] = v; lsum += v; lsq += v * v;
  }
  __shared__ float s1[256], s2[256];
  s1[threadIdx.x] = lsum; s2[threadIdx.x] = lsq;
  __syncthreads();
  for (int off = 128; off > 0; off >>= 1) {
    if (threadIdx.x < off) {
      s1[threadIdx.x] += s1[threadIdx.x + off];
      s2[threadIdx.x] += s2[threadIdx.x + off];
    }
    __syncthreads();
  }
  float mu = s1[0] * (1.f / 1024.f);
  float var = s2[0] * (1.f / 1024.f) - mu * mu;
  float rinv = rsqrtf(var + 1e-5f);
#pragma unroll
  for (int i = 0; i < 4; ++i) {
    int d = threadIdx.x + 256 * i;
    op[d] = (_Float16)((vals[i] - mu) * rinv * w[d] + b[d]);
  }
}

// ---------------------------------------------------------------------------
// Kernel 3: fused attention, one workgroup per (head, batch).
// 128 threads = 4 waves; wave w owns M-tile row w (A frag hoisted, reused x4).
// Staging uses global_load_async_to_lds_b128 (ASYNCcnt) when available.
// ---------------------------------------------------------------------------
__device__ __forceinline__ void stage_w(_Float16* Ws, const _Float16* __restrict__ W, int tid) {
#if HAVE_ASYNC_LDS
#pragma unroll
  for (int i = 0; i < 4; ++i) {
    int idx = tid + i * 128;  // 512 x 16B chunks = 4096 halves
    async_copy16(W + idx * 8, Ws + idx * 8);
  }
#else
  for (int idx = tid; idx < 4096; idx += 128) Ws[idx] = W[idx];
#endif
}

// MODE: 0 -> Q row-major [s][e]; 1 -> K row-major [j][e] (B_t for scores);
//       2 -> V transposed [e][j] (B_t for P@V)
template <int MODE>
__device__ __forceinline__ void qkv_gemm(const _Float16* Xs, const _Float16* Ws,
                                         const float* __restrict__ bias,
                                         _Float16* dst, int wv, int hl, int lm) {
  v8f acc[4] = {};
#pragma unroll
  for (int kk = 0; kk < 64; kk += 32) {
    v16h af = load_a_frag(Xs + wv * 16 * 64 + kk, 64);
#pragma unroll
    for (int nt = 0; nt < 4; ++nt) {
      v16h bf = load_b_frag_t(Ws + nt * 16 * 64 + kk, 64);
      acc[nt] = WMMA_F16(af, bf, acc[nt]);
    }
  }
#pragma unroll
  for (int nt = 0; nt < 4; ++nt) {
#pragma unroll
    for (int r = 0; r < 8; ++r) {
      int row = wv * 16 + r + 8 * hl;
      int col = nt * 16 + lm;
      float val = acc[nt][r] + bias[col];
      if (MODE == 2) dst[col * 64 + row] = (_Float16)val;
      else           dst[row * 64 + col] = (_Float16)val;
    }
  }
}

__global__ __launch_bounds__(128) void attn_kernel(const _Float16* __restrict__ nh,
                                                   const _Float16* __restrict__ wqh, const float* __restrict__ bq,
                                                   const _Float16* __restrict__ wkh, const float* __restrict__ bk,
                                                   const _Float16* __restrict__ wvh, const float* __restrict__ bv,
                                                   float* __restrict__ tok) {
  int h = blockIdx.x, bb = blockIdx.y;
  int tid = threadIdx.x;
  int lane = tid & 31, wv = tid >> 5;
  int lm = lane & 15, hl = lane >> 4;

  __shared__ __attribute__((aligned(32))) _Float16 Xs[64 * 64];  // LN1 slice; later P
  __shared__ __attribute__((aligned(32))) _Float16 Ws[64 * 64];  // W[h] staged (B_t)
  __shared__ __attribute__((aligned(32))) _Float16 Qs[64 * 64];  // [s][e]
  __shared__ __attribute__((aligned(32))) _Float16 Ks[64 * 64];  // [j][e] == B_t for scores
  __shared__ __attribute__((aligned(32))) _Float16 Vt[64 * 64];  // [e][j] == B_t for P@V
  __shared__ float Ss[64 * 64];                                  // scores f32

  // Stage X = nh[bb, :, h*64 : h*64+64]; zero-pad rows 50..63
#if HAVE_ASYNC_LDS
#pragma unroll
  for (int i = 0; i < 4; ++i) {
    int idx = tid + i * 128;  // 16B chunk index; 8 chunks per 64-half row
    if (idx < 400) {          // 50 rows * 8 chunks
      int s = idx >> 3, c = (idx & 7) * 8;
      async_copy16(nh + ((long)(bb * 50 + s) << 10) + h * 64 + c, Xs + idx * 8);
    } else if (idx < 512) {
      v8h z = {};
      *(v8h*)(Xs + idx * 8) = z;
    }
  }
#else
  for (int idx = tid; idx < 64 * 64; idx += 128) {
    int s = idx >> 6, d = idx & 63;
    Xs[idx] = (s < 50) ? nh[((long)(bb * 50 + s) << 10) + h * 64 + d] : (_Float16)0.f;
  }
#endif

  // ---- Q = X @ Wq^T + bq  (B_t[e][d] = Wq[h][e][d] : direct row-major copy)
  stage_w(Ws, wqh + h * 4096, tid);
  async_wait0();
  __syncthreads();
  qkv_gemm<0>(Xs, Ws, bq + h * 64, Qs, wv, hl, lm);
  __syncthreads();
  // ---- K
  stage_w(Ws, wkh + h * 4096, tid);
  async_wait0();
  __syncthreads();
  qkv_gemm<1>(Xs, Ws, bk + h * 64, Ks, wv, hl, lm);
  __syncthreads();
  // ---- V (stored transposed)
  stage_w(Ws, wvh + h * 4096, tid);
  async_wait0();
  __syncthreads();
  qkv_gemm<2>(Xs, Ws, bv + h * 64, Vt, wv, hl, lm);
  __syncthreads();

  // ---- scores = (Q @ K^T) * sqrt(1024)
  {
    v8f acc[4] = {};
#pragma unroll
    for (int kk = 0; kk < 64; kk += 32) {
      v16h af = load_a_frag(Qs + wv * 16 * 64 + kk, 64);
#pragma unroll
      for (int nt = 0; nt < 4; ++nt) {
        v16h bf = load_b_frag_t(Ks + nt * 16 * 64 + kk, 64);
        acc[nt] = WMMA_F16(af, bf, acc[nt]);
      }
    }
#pragma unroll
    for (int nt = 0; nt < 4; ++nt)
#pragma unroll
      for (int r = 0; r < 8; ++r)
        Ss[(wv * 16 + r + 8 * hl) * 64 + nt * 16 + lm] = acc[nt][r] * 32.0f;
  }
  __syncthreads();

  // ---- row softmax (cols 0..49 valid; pad -> 0); P overwrites Xs
  _Float16* Ps = Xs;
  if (tid < 64) {
    int i = tid;
    if (i < 50) {
      float mx = -1e30f;
      for (int j = 0; j < 50; ++j) mx = fmaxf(mx, Ss[i * 64 + j]);
      float sum = 0.f;
      for (int j = 0; j < 50; ++j) {
        float e = __expf(Ss[i * 64 + j] - mx);
        Ss[i * 64 + j] = e;
        sum += e;
      }
      float rs = 1.f / sum;
      for (int j = 0; j < 64; ++j)
        Ps[i * 64 + j] = (j < 50) ? (_Float16)(Ss[i * 64 + j] * rs) : (_Float16)0.f;
    } else {
      for (int j = 0; j < 64; ++j) Ps[i * 64 + j] = (_Float16)0.f;
    }
  }
  __syncthreads();

  // ---- O = P @ V; accumulate into tok (disjoint head column slice)
  {
    v8f acc[4] = {};
#pragma unroll
    for (int kk = 0; kk < 64; kk += 32) {
      v16h af = load_a_frag(Ps + wv * 16 * 64 + kk, 64);
#pragma unroll
      for (int nt = 0; nt < 4; ++nt) {
        v16h bf = load_b_frag_t(Vt + nt * 16 * 64 + kk, 64);
        acc[nt] = WMMA_F16(af, bf, acc[nt]);
      }
    }
#pragma unroll
    for (int nt = 0; nt < 4; ++nt)
#pragma unroll
      for (int r = 0; r < 8; ++r) {
        int s = wv * 16 + r + 8 * hl;
        if (s < 50) {
          int e = nt * 16 + lm;
          long gi = ((long)(bb * 50 + s) << 10) + h * 64 + e;
          tok[gi] += acc[nt][r];
        }
      }
  }
}

// ---------------------------------------------------------------------------
// Shared GEMM staging: copy one 64x32 f16 tile (row-stride 1024) into LDS.
// idx = 16B chunk: row = idx>>2, col = (idx&3)*8
// ---------------------------------------------------------------------------
__device__ __forceinline__ void stage_tile(_Float16* Ls, const _Float16* __restrict__ G,
                                           long row0, int k0, int tid) {
#if HAVE_ASYNC_LDS
#pragma unroll
  for (int i = 0; i < 2; ++i) {
    int idx = tid + i * 128;  // 256 chunks
    int row = idx >> 2, c = (idx & 3) * 8;
    async_copy16(G + (row0 + row) * 1024 + k0 + c, Ls + idx * 8);
  }
#else
  for (int idx = tid; idx < 64 * 32; idx += 128) {
    int i = idx >> 5, j = idx & 31;
    Ls[idx] = G[(row0 + i) * 1024 + k0 + j];
  }
#endif
}

// ---------------------------------------------------------------------------
// Kernel 4: class-token MLP GEMM: cls = tok[:,0] + LN2(tok[:,0]) @ Wm^T + bm
// M=N=K=1024, f16 output for the head GEMM. grid (16,16), 128 threads.
// ---------------------------------------------------------------------------
__global__ __launch_bounds__(128) void mlp_cls_kernel(const _Float16* __restrict__ Aln,
                                                      const _Float16* __restrict__ wmh,
                                                      const float* __restrict__ bmv,
                                                      const float* __restrict__ tok,
                                                      _Float16* __restrict__ clsh) {
  int tnb = blockIdx.x, tmb = blockIdx.y;
  int tid = threadIdx.x, lane = tid & 31, wv = tid >> 5;
  int lm = lane & 15, hl = lane >> 4;
  __shared__ __attribute__((aligned(32))) _Float16 As[64 * 32];  // [m][k]
  __shared__ __attribute__((aligned(32))) _Float16 Bs[64 * 32];  // [n][k]  (B_t)
  v8f acc[4] = {};
  for (int k0 = 0; k0 < 1024; k0 += 32) {
    __syncthreads();
    stage_tile(As, Aln, (long)tmb * 64, k0, tid);
    stage_tile(Bs, wmh, (long)tnb * 64, k0, tid);
    async_wait0();
    __syncthreads();
    v16h af = load_a_frag(As + wv * 16 * 32, 32);
#pragma unroll
    for (int nt = 0; nt < 4; ++nt) {
      v16h bf = load_b_frag_t(Bs + nt * 16 * 32, 32);
      acc[nt] = WMMA_F16(af, bf, acc[nt]);
    }
  }
#pragma unroll
  for (int nt = 0; nt < 4; ++nt)
#pragma unroll
    for (int r = 0; r < 8; ++r) {
      int row = tmb * 64 + wv * 16 + r + 8 * hl;  // batch index
      int col = tnb * 64 + nt * 16 + lm;
      clsh[(long)row * 1024 + col] =
          (_Float16)(acc[nt][r] + bmv[col] + tok[(long)row * 51200 + col]);
    }
}

// ---------------------------------------------------------------------------
// Kernel 5: head GEMM: logits = cls @ Wh^T + bh  (Wh pre-padded to 1024 rows)
// ---------------------------------------------------------------------------
__global__ __launch_bounds__(128) void head_kernel(const _Float16* __restrict__ clsh,
                                                   const _Float16* __restrict__ whh,
                                                   const float* __restrict__ bhv,
                                                   float* __restrict__ logits) {
  int tnb = blockIdx.x, tmb = blockIdx.y;
  int tid = threadIdx.x, lane = tid & 31, wv = tid >> 5;
  int lm = lane & 15, hl = lane >> 4;
  __shared__ __attribute__((aligned(32))) _Float16 As[64 * 32];
  __shared__ __attribute__((aligned(32))) _Float16 Bs[64 * 32];  // [n][k]
  v8f acc[4] = {};
  for (int k0 = 0; k0 < 1024; k0 += 32) {
    __syncthreads();
    stage_tile(As, clsh, (long)tmb * 64, k0, tid);
    stage_tile(Bs, whh, (long)tnb * 64, k0, tid);
    async_wait0();
    __syncthreads();
    v16h af = load_a_frag(As + wv * 16 * 32, 32);
#pragma unroll
    for (int nt = 0; nt < 4; ++nt) {
      v16h bf = load_b_frag_t(Bs + nt * 16 * 32, 32);
      acc[nt] = WMMA_F16(af, bf, acc[nt]);
    }
  }
#pragma unroll
  for (int nt = 0; nt < 4; ++nt)
#pragma unroll
    for (int r = 0; r < 8; ++r) {
      int row = tmb * 64 + wv * 16 + r + 8 * hl;
      int col = tnb * 64 + nt * 16 + lm;
      if (col < 1000) logits[(long)row * 1024 + col] = acc[nt][r] + bhv[col];
    }
}

// ---------------------------------------------------------------------------
// Kernel 6: row softmax over 1000 logits -> d_out
// ---------------------------------------------------------------------------
__global__ __launch_bounds__(256) void softmax_kernel(const float* __restrict__ logits,
                                                      float* __restrict__ out) {
  int bb = blockIdx.x;
  const float* lp = logits + (long)bb * 1024;
  __shared__ float red[256];
  float mx = -1e30f;
  for (int o = threadIdx.x; o < 1000; o += 256) mx = fmaxf(mx, lp[o]);
  red[threadIdx.x] = mx;
  __syncthreads();
  for (int off = 128; off > 0; off >>= 1) {
    if (threadIdx.x < off) red[threadIdx.x] = fmaxf(red[threadIdx.x], red[threadIdx.x + off]);
    __syncthreads();
  }
  mx = red[0];
  __syncthreads();
  float sum = 0.f;
  for (int o = threadIdx.x; o < 1000; o += 256) sum += __expf(lp[o] - mx);
  red[threadIdx.x] = sum;
  __syncthreads();
  for (int off = 128; off > 0; off >>= 1) {
    if (threadIdx.x < off) red[threadIdx.x] += red[threadIdx.x + off];
    __syncthreads();
  }
  float rs = 1.f / red[0];
  for (int o = threadIdx.x; o < 1000; o += 256) out[(long)bb * 1000 + o] = __expf(lp[o] - mx) * rs;
}

// ---------------------------------------------------------------------------
extern "C" void kernel_launch(void* const* d_in, const int* in_sizes, int n_in,
                              void* d_out, int out_size, void* d_ws, size_t ws_size,
                              hipStream_t stream) {
  const float* x    = (const float*)d_in[0];
  const float* cls  = (const float*)d_in[1];
  const float* Wp   = (const float*)d_in[2];
  const float* ln1w = (const float*)d_in[3];
  const float* ln1b = (const float*)d_in[4];
  const float* Wq   = (const float*)d_in[5];
  const float* bq   = (const float*)d_in[6];
  const float* Wk   = (const float*)d_in[7];
  const float* bk   = (const float*)d_in[8];
  const float* Wv   = (const float*)d_in[9];
  const float* bv   = (const float*)d_in[10];
  const float* ln2w = (const float*)d_in[11];
  const float* ln2b = (const float*)d_in[12];
  const float* Wm   = (const float*)d_in[13];
  const float* bmv  = (const float*)d_in[14];
  const float* Wh   = (const float*)d_in[15];
  const float* bhv  = (const float*)d_in[16];

  char* ws = (char*)d_ws;
  float*    tok    = (float*)ws;                        // 200 MB
  _Float16* nh     = (_Float16*)(ws + 209715200L);      // 100 MB
  _Float16* clsln  = (_Float16*)(ws + 314572800L);      // 2 MB
  _Float16* clsh   = (_Float16*)(ws + 316669952L);      // 2 MB
  float*    logits = (float*)(ws + 318767104L);         // 4 MB
  _Float16* wqh    = (_Float16*)(ws + 322961408L);      // 128 KB
  _Float16* wkh    = (_Float16*)(ws + 323092480L);      // 128 KB
  _Float16* wvh    = (_Float16*)(ws + 323223552L);      // 128 KB
  _Float16* wmh    = (_Float16*)(ws + 323354624L);      // 2 MB
  _Float16* whh    = (_Float16*)(ws + 325451776L);      // 2 MB (padded 1024x1024)
  float*    outp   = (float*)d_out;

  // one-time weight conversions (f32 -> f16) for async LDS staging
  cvt_f16_kernel<<<256, 256, 0, stream>>>(Wq, wqh, 65536);
  cvt_f16_kernel<<<256, 256, 0, stream>>>(Wk, wkh, 65536);
  cvt_f16_kernel<<<256, 256, 0, stream>>>(Wv, wvh, 65536);
  cvt_f16_kernel<<<4096, 256, 0, stream>>>(Wm, wmh, 1048576);
  cvt_pad_wh_kernel<<<4096, 256, 0, stream>>>(Wh, whh);

  embed_kernel<<<dim3(50, 1024), 256, 0, stream>>>(x, cls, Wp, tok);
  ln_kernel<<<51200, 256, 0, stream>>>(tok, 1024, ln1w, ln1b, nh, 1024);
  attn_kernel<<<dim3(16, 1024), 128, 0, stream>>>(nh, wqh, bq, wkh, bk, wvh, bv, tok);
  ln_kernel<<<1024, 256, 0, stream>>>(tok, 51200, ln2w, ln2b, clsln, 1024);  // class rows
  mlp_cls_kernel<<<dim3(16, 16), 128, 0, stream>>>(clsln, wmh, bmv, tok, clsh);
  head_kernel<<<dim3(16, 16), 128, 0, stream>>>(clsh, whh, bhv, logits);
  softmax_kernel<<<1024, 256, 0, stream>>>(logits, outp);
}
